// FNP_RGCNPredictor_82446192214117
// MI455X (gfx1250) — compile-verified, matrix operand order checked by reading, not run on hardware
//
#include <hip/hip_runtime.h>
#include <stdint.h>

#define NN 50000
#define NE 200000
#define DIN 128
#define DD 256
#define NL 4
#define NR 4
#define NM 1024
#define NRAT 64

typedef __attribute__((ext_vector_type(16))) __bf16 v16bf;
typedef __attribute__((ext_vector_type(8)))  float  v8f;

union FragBF { v16bf v; uint4 q[2]; };

__host__ __device__ static inline uint32_t rotl32(uint32_t x, int r){ return (x<<r)|(x>>(32-r)); }

// Threefry-2x32, 20 rounds (JAX PRNG core)
__host__ __device__ static inline void threefry2x32(uint32_t k0, uint32_t k1,
                                                    uint32_t x0, uint32_t x1,
                                                    uint32_t* o0, uint32_t* o1){
  const int R[8] = {13,15,26,6,17,29,16,24};
  uint32_t ks[3] = {k0, k1, k0 ^ k1 ^ 0x1BD11BDAu};
  x0 += ks[0]; x1 += ks[1];
  for (int g=0; g<5; ++g){
    for (int j=0; j<4; ++j){
      x0 += x1; x1 = rotl32(x1, R[(g&1)*4+j]); x1 ^= x0;
    }
    x0 += ks[(g+1)%3];
    x1 += ks[(g+2)%3] + (uint32_t)(g+1);
  }
  *o0 = x0; *o1 = x1;
}

__device__ static inline uint32_t jax_bits(uint32_t k0,uint32_t k1,uint32_t idx,uint32_t n){
  uint32_t h = n >> 1, o0, o1;
  if (idx < h){ threefry2x32(k0,k1, idx, idx+h, &o0,&o1); return o0; }
  threefry2x32(k0,k1, idx-h, idx, &o0,&o1); return o1;
}
__device__ static inline float jax_unit(uint32_t k0,uint32_t k1,uint32_t idx,uint32_t n){
  uint32_t b = jax_bits(k0,k1,idx,n);
  return __uint_as_float((b>>9)|0x3F800000u) - 1.0f;  // [0,1)
}
__device__ static inline float jax_normal(uint32_t k0,uint32_t k1,uint32_t idx,uint32_t n){
  float u = jax_unit(k0,k1,idx,n)*2.0f - 1.0f;
  u = fminf(fmaxf(u, -0.99999994f), 0.99999994f);
  return 1.41421356237f * erfinvf(u);
}

__device__ static inline uint16_t f2bf(float f){
  uint32_t u = __float_as_uint(f);
  uint32_t r = u + 0x7FFFu + ((u>>16)&1u);
  if ((u & 0x7F800000u) == 0x7F800000u) r = u;   // inf/nan passthrough
  return (uint16_t)(r>>16);
}

// ---------------------------------------------------------------------------
// Async global->LDS stage of one 128x32 bf16 B-tile (8KB = 512 x 16B chunks,
// 2 chunks per thread at 256 threads). Uses gfx1250 async-DMA path, tracked
// by ASYNCcnt.
// ---------------------------------------------------------------------------
__device__ static inline void issue_b_stage(const uint16_t* __restrict__ Bt,
                                            int nbase, int K, int k0,
                                            uint16_t* buf){
  #pragma unroll
  for (int j=0; j<2; ++j){
    int c = (int)threadIdx.x + j*256;   // chunk id 0..511
    int row = c >> 2, cc = c & 3;       // 4 x 16B chunks per 32-elem row
    const uint16_t* g = Bt + (size_t)(nbase + row)*K + k0 + cc*8;
    uint32_t lds = (uint32_t)(size_t)(buf + row*32 + cc*8);
    unsigned long long ga = (unsigned long long)(size_t)g;
    asm volatile("global_load_async_to_lds_b128 %0, %1, off"
                 :: "v"(lds), "v"(ga) : "memory");
  }
}

// ---------------------------------------------------------------------------
// Generic WMMA GEMM: outF[M,N] = A_bf16[M,K] * Bt_bf16[N,K]^T (+C0 +bias), relu opt.
// Block: 256 thr = 8 waves; block tile 64x128; wave tile 16x64 (4x v8f acc).
// B tiles double-buffered in LDS via global_load_async_to_lds_b128.
// ---------------------------------------------------------------------------
__global__ __launch_bounds__(256) void wmma_gemm_bf16(
    const uint16_t* __restrict__ A, const uint16_t* __restrict__ Bt,
    const float* __restrict__ C0, const float* __restrict__ bias,
    float* __restrict__ outF, uint16_t* __restrict__ outB,
    int M, int N, int K, int relu)
{
  __shared__ uint16_t sB[2][128*32];   // 2 x 8KB double buffer

  const int lane = threadIdx.x & 31;
  const int wave = threadIdx.x >> 5;
  const int lo = lane & 15, hi = lane >> 4;
  const int wr = wave & 3, wc = wave >> 2;
  const int row0 = blockIdx.y*64 + wr*16;
  const int col0 = blockIdx.x*128 + wc*64;
  const int nbase = blockIdx.x*128;

  v8f acc[4];
  #pragma unroll
  for (int t=0;t<4;++t){
    #pragma unroll
    for (int i=0;i<8;++i){
      int r = row0 + hi*8 + i;
      int c = col0 + t*16 + lo;
      float v = 0.0f;
      if (C0 && r < M) v = C0[(size_t)r*N + c];
      if (bias) v += bias[c];
      acc[t][i] = v;
    }
  }

  const int arow = row0 + lo;

  // prologue: start DMA of first B stage
  issue_b_stage(Bt, nbase, K, 0, sB[0]);

  for (int k0 = 0; k0 < K; k0 += 32){
    const int cur = (k0 >> 5) & 1;

    // current stage's async loads complete, then make visible block-wide
    asm volatile("s_wait_asynccnt 0" ::: "memory");
    __syncthreads();

    // kick off DMA of the next stage into the other buffer (its previous
    // readers all finished before the barrier above)
    if (k0 + 32 < K) issue_b_stage(Bt, nbase, K, k0 + 32, sB[cur ^ 1]);

    // A fragment streamed from global (+ prefetch one K-step ahead)
    FragBF a;
    if (arow < M){
      const uint16_t* p = A + (size_t)arow*K + k0 + hi*8;
      a.q[0] = *(const uint4*)(p);
      a.q[1] = *(const uint4*)(p + 16);
      if (k0 + 32 < K) __builtin_prefetch(p + 32, 0, 1);
    } else {
      a.q[0] = make_uint4(0,0,0,0);
      a.q[1] = make_uint4(0,0,0,0);
    }

    const uint16_t* bbuf = &sB[cur][0];
    #pragma unroll
    for (int t=0;t<4;++t){
      const uint16_t* p = bbuf + (size_t)(wc*64 + t*16 + lo)*32 + hi*8;
      FragBF b;
      b.q[0] = *(const uint4*)(p);        // K = hi*8 + 0..7
      b.q[1] = *(const uint4*)(p + 16);   // K = 16 + hi*8 + 0..7
      acc[t] = __builtin_amdgcn_wmma_f32_16x16x32_bf16(false, a.v, false, b.v,
                                                       (short)0, acc[t], false, false);
    }
  }

  #pragma unroll
  for (int t=0;t<4;++t){
    #pragma unroll
    for (int i=0;i<8;++i){
      int r = row0 + hi*8 + i;
      if (r >= M) continue;
      int c = col0 + t*16 + lo;
      float v = acc[t][i];
      if (relu) v = fmaxf(v, 0.0f);
      outF[(size_t)r*N + c] = v;
      if (outB) outB[(size_t)r*N + c] = f2bf(v);
    }
  }
}

// ---------------------------------------------------------------------------
// Utility kernels
// ---------------------------------------------------------------------------
__global__ void zero_f32(float* p, int n){
  int i = blockIdx.x*blockDim.x + threadIdx.x;
  if (i < n) p[i] = 0.0f;
}

__global__ void conv_f2b(const float* __restrict__ src, uint16_t* __restrict__ dst, int n){
  int i = blockIdx.x*blockDim.x + threadIdx.x;
  if (i < n) dst[i] = f2bf(src[i]);
}

// transpose-convert f32 [K,N] -> bf16 [N,K]
__global__ void convT_f2b(const float* __restrict__ src, uint16_t* __restrict__ dst, int K, int N){
  int idx = blockIdx.x*blockDim.x + threadIdx.x;
  if (idx >= K*N) return;
  int nn = idx / K, kk = idx - nn*K;
  dst[idx] = f2bf(src[(size_t)kk*N + nn]);
}

__global__ void count_edges(const int* __restrict__ ei, const int* __restrict__ et,
                            float* __restrict__ cnt){
  int e = blockIdx.x*blockDim.x + threadIdx.x;
  if (e >= NE) return;
  int dst = ei[NE + e];
  atomicAdd(&cnt[(size_t)et[e]*NN + dst], 1.0f);
}

__global__ void scatter_rel(const int* __restrict__ ei, const int* __restrict__ et,
                            const float* __restrict__ cnt, const float* __restrict__ hrel,
                            float* __restrict__ agg, int rel){
  int e = blockIdx.x;
  if (et[e] != rel) return;
  int src = ei[e], dst = ei[NE + e];
  float inv = 1.0f / fmaxf(cnt[(size_t)rel*NN + dst], 1.0f);
  int c = threadIdx.x;
  atomicAdd(&agg[(size_t)dst*DD + c], hrel[(size_t)src*DD + c] * inv);
}

__global__ void pool_sum(const float* __restrict__ feats, const int* __restrict__ batch,
                         float* __restrict__ hm, float* __restrict__ gcnt){
  int v = blockIdx.x, c = threadIdx.x;
  int g = batch[v];
  atomicAdd(&hm[(size_t)g*DD + c], feats[(size_t)v*DD + c]);
  if (c == 0) atomicAdd(&gcnt[g], 1.0f);
}

__global__ void pool_div(float* hm, const float* gcnt){
  int idx = blockIdx.x*blockDim.x + threadIdx.x;
  if (idx >= NM*DD) return;
  int g = idx >> 8;
  hm[idx] = hm[idx] / fmaxf(gcnt[g], 1.0f);
}

__global__ void hall_relu(const float* __restrict__ HR, const float* __restrict__ hm,
                          uint16_t* __restrict__ out){
  int idx = blockIdx.x*blockDim.x + threadIdx.x;
  if (idx >= (NRAT+NM)*DD) return;
  int row = idx >> 8, c = idx & 255;
  float v = (row < NRAT) ? HR[(size_t)row*DD + c] : hm[(size_t)(row-NRAT)*DD + c];
  out[idx] = f2bf(fmaxf(v, 0.0f));
}

__global__ void sample_u(const float* __restrict__ pu, float* __restrict__ u,
                         uint32_t k0, uint32_t k1){
  int idx = blockIdx.x*blockDim.x + threadIdx.x;
  const int total = (NRAT+NM)*DD;
  if (idx >= total) return;
  int row = idx >> 8, c = idx & 255;
  float mean = pu[(size_t)row*512 + c];
  float ls   = pu[(size_t)row*512 + 256 + c];
  u[idx] = mean + expf(ls) * jax_normal(k0,k1,(uint32_t)idx,(uint32_t)total);
}

__global__ void sample_z(const float* __restrict__ qz, float* __restrict__ z,
                         uint32_t k0, uint32_t k1, int rows, int rowoff){
  int idx = blockIdx.x*blockDim.x + threadIdx.x;
  if (idx >= rows*DD) return;
  int row = idx >> 8, c = idx & 255;
  float mean = qz[(size_t)(rowoff+row)*512 + c];
  float ls   = qz[(size_t)(rowoff+row)*512 + 256 + c];
  z[idx] = mean + expf(ls) * jax_normal(k0,k1,(uint32_t)idx,(uint32_t)(rows*DD));
}

__global__ void compute_An(const float* __restrict__ u, const float* __restrict__ pg_logscale,
                           uint32_t ka0, uint32_t ka1, float* __restrict__ An){
  int m = blockIdx.x, r = threadIdx.x;  // block = 64 threads
  const float* uM = u + (size_t)(NRAT+m)*DD;
  const float* uR = u + (size_t)r*DD;
  float d2 = 0.0f;
  for (int d=0; d<DD; ++d){ float t = uM[d]-uR[d]; d2 += t*t; }
  float gscale = expf(pg_logscale[0]);
  float logp = -0.5f * d2 / gscale;
  float em = fmaxf(-expm1f(logp), 1e-20f);
  float logits = logp - logf(em);
  float uu = jax_unit(ka0,ka1,(uint32_t)(m*NRAT + r),(uint32_t)(NM*NRAT));
  uu = 1e-6f + uu * (1.0f - 2e-6f);
  float g = (logits + logf(uu) - log1pf(-uu)) / 0.3f;
  float A = 1.0f / (1.0f + expf(-g));
  __shared__ float sA[NRAT];
  __shared__ float ssum;
  sA[r] = A; __syncthreads();
  if (r == 0){ float s = 0.0f; for (int i=0;i<NRAT;++i) s += sA[i]; ssum = s; }
  __syncthreads();
  An[(size_t)m*NRAT + r] = A / (ssum + 1e-8f);
}

__global__ void pz_logpqz(const float* __restrict__ An, const float* __restrict__ qz,
                          const float* __restrict__ zM, float* __restrict__ accum){
  int m = blockIdx.x, d = threadIdx.x;  // block = 256 threads
  float pzm = 0.0f, pzl = 0.0f;
  for (int r=0; r<NRAT; ++r){
    float a = An[(size_t)m*NRAT + r];
    pzm += a * qz[(size_t)r*512 + d];
    pzl += a * qz[(size_t)r*512 + 256 + d];
  }
  float z  = zM[(size_t)m*DD + d];
  float qm = qz[(size_t)(NRAT+m)*512 + d];
  float ql = qz[(size_t)(NRAT+m)*512 + 256 + d];
  float t1 = (z - pzm) * expf(-pzl);
  float t2 = (z - qm)  * expf(-ql);
  float val = (ql - pzl) - 0.5f*t1*t1 + 0.5f*t2*t2;
  __shared__ float s[DD];
  s[d] = val; __syncthreads();
  for (int st=128; st>0; st>>=1){ if (d < st) s[d] += s[d+st]; __syncthreads(); }
  if (d == 0) atomicAdd(accum, s[0]);
}

__global__ void build_rep(const float* __restrict__ z, const float* __restrict__ u,
                          int uoff, uint16_t* __restrict__ rep, int rows){
  int idx = blockIdx.x*blockDim.x + threadIdx.x;
  if (idx >= rows*512) return;
  int row = idx >> 9, c = idx & 511;
  float v = (c < DD) ? z[(size_t)row*DD + c] : u[(size_t)(uoff+row)*DD + (c-DD)];
  rep[idx] = f2bf(v);
}

__global__ void head_loss(const float* __restrict__ hid, const float* __restrict__ W2,
                          const float* __restrict__ b2, const int* __restrict__ y,
                          float* __restrict__ accum){
  int m = blockIdx.x, k = threadIdx.x;  // block = 256
  float h = hid[(size_t)m*DD + k];
  __shared__ float s0[DD], s1[DD];
  s0[k] = h * W2[k*2 + 0];
  s1[k] = h * W2[k*2 + 1];
  __syncthreads();
  for (int st=128; st>0; st>>=1){
    if (k < st){ s0[k] += s0[k+st]; s1[k] += s1[k+st]; }
    __syncthreads();
  }
  if (k == 0){
    float l0 = s0[0] + b2[0], l1 = s1[0] + b2[1];
    float mx = fmaxf(l0, l1);
    float lse = mx + logf(expf(l0-mx) + expf(l1-mx));
    float lp = ((y[m] == 0) ? l0 : l1) - lse;
    atomicAdd(accum, lp);
  }
}

__global__ void reg_kernel(const float* __restrict__ u, const int* __restrict__ yM,
                           float* __restrict__ accum){
  int m = blockIdx.x, i = threadIdx.x;  // block = 32
  int idx = i + (1 - yM[m]) * (NRAT/2);
  const float* ur = u + (size_t)idx*DD;
  const float* um = u + (size_t)(NRAT+m)*DD;
  float d2 = 0.0f;
  for (int d=0; d<DD; ++d){ float t = ur[d]-um[d]; d2 += t*t; }
  float pd = sqrtf(fmaxf(d2, 1e-12f));
  __shared__ float s[32];
  s[i] = pd; __syncthreads();
  for (int st=16; st>0; st>>=1){ if (i < st) s[i] += s[i+st]; __syncthreads(); }
  if (i == 0) atomicAdd(accum, s[0]);
}

__global__ void finalize(const float* __restrict__ scal, float* __restrict__ out){
  if (threadIdx.x == 0 && blockIdx.x == 0){
    float pred = -(scal[0] + 0.1f * scal[3]) / (float)NM;
    float rat  = -scal[1] / (float)NRAT;
    float reg  = scal[2] / (float)((NRAT/2) * NM);
    out[0] = pred + rat - 0.1f * reg;
  }
}

// ---------------------------------------------------------------------------
extern "C" void kernel_launch(void* const* d_in, const int* in_sizes, int n_in,
                              void* d_out, int out_size, void* d_ws, size_t ws_size,
                              hipStream_t stream)
{
  const float* x          = (const float*)d_in[0];
  const int*   edge_index = (const int*)  d_in[1];
  const int*   edge_type  = (const int*)  d_in[2];
  const int*   batch      = (const int*)  d_in[3];
  const int*   yM         = (const int*)  d_in[4];
  const int*   yR         = (const int*)  d_in[5];
  const float* H_R        = (const float*)d_in[6];
  const float* node_W     = (const float*)d_in[7];
  const float* node_b     = (const float*)d_in[8];
  const float* gcn_Wr     = (const float*)d_in[9];
  const float* gcn_Wroot  = (const float*)d_in[10];
  const float* gcn_b      = (const float*)d_in[11];
  const float* pg_ls      = (const float*)d_in[12];
  const float* pu_W       = (const float*)d_in[13];
  const float* pu_b       = (const float*)d_in[14];
  const float* qz_W       = (const float*)d_in[15];
  const float* qz_b       = (const float*)d_in[16];
  const float* out_W1     = (const float*)d_in[17];
  const float* out_b1     = (const float*)d_in[18];
  const float* out_W2     = (const float*)d_in[19];
  const float* out_b2     = (const float*)d_in[20];
  float* out = (float*)d_out;

  // workspace bump allocator
  char* base = (char*)d_ws; size_t off = 0;
  auto alloc = [&](size_t bytes)->char*{
    char* p = base + off; off += (bytes + 255) & ~(size_t)255; return p;
  };
  uint16_t* xb      = (uint16_t*)alloc((size_t)NN*DIN*2);
  uint16_t* featsb  = (uint16_t*)alloc((size_t)NN*DD*2);
  float*    feats   = (float*)   alloc((size_t)NN*DD*4);
  float*    hrel    = (float*)   alloc((size_t)NN*DD*4);
  float*    agg     = (float*)   alloc((size_t)NN*DD*4);
  float*    cnt     = (float*)   alloc((size_t)NR*NN*4);
  uint16_t* wtnode  = (uint16_t*)alloc((size_t)DD*DIN*2);
  uint16_t* wtWr    = (uint16_t*)alloc((size_t)NL*NR*DD*DD*2);
  uint16_t* wtWroot = (uint16_t*)alloc((size_t)NL*DD*DD*2);
  uint16_t* wtpu    = (uint16_t*)alloc((size_t)512*DD*2);
  uint16_t* wtqz    = (uint16_t*)alloc((size_t)512*DD*2);
  uint16_t* wto1    = (uint16_t*)alloc((size_t)DD*512*2);
  float*    hm      = (float*)   alloc((size_t)NM*DD*4);
  float*    gcnt    = (float*)   alloc((size_t)NM*4);
  uint16_t* hallb   = (uint16_t*)alloc((size_t)(NRAT+NM)*DD*2);
  float*    pu_out  = (float*)   alloc((size_t)(NRAT+NM)*512*4);
  float*    qz_out  = (float*)   alloc((size_t)(NRAT+NM)*512*4);
  float*    u       = (float*)   alloc((size_t)(NRAT+NM)*DD*4);
  float*    An      = (float*)   alloc((size_t)NM*NRAT*4);
  float*    zM      = (float*)   alloc((size_t)NM*DD*4);
  float*    zR      = (float*)   alloc((size_t)NRAT*DD*4);
  uint16_t* repMb   = (uint16_t*)alloc((size_t)NM*512*2);
  uint16_t* repRb   = (uint16_t*)alloc((size_t)NRAT*512*2);
  float*    hidM    = (float*)   alloc((size_t)NM*DD*4);
  float*    hidR    = (float*)   alloc((size_t)NRAT*DD*4);
  float*    scal    = (float*)   alloc(4*4);

  // JAX subkeys: key(42) = (0,42); split(key,4) via threefry on iota(8) halves
  uint32_t ku0,ku1,ka0,ka1,kzm0,kzm1,kzr0,kzr1, o0,o1;
  threefry2x32(0u,42u,0u,4u,&o0,&o1); ku0=o0;  ku1=o1;
  threefry2x32(0u,42u,1u,5u,&o0,&o1); ka0=o0;  ka1=o1;
  threefry2x32(0u,42u,2u,6u,&o0,&o1); kzm0=o0; kzm1=o1;
  threefry2x32(0u,42u,3u,7u,&o0,&o1); kzr0=o0; kzr1=o1;

  auto gemm = [&](const uint16_t* A, const uint16_t* Bt, const float* C0,
                  const float* bias, float* outF, uint16_t* outB,
                  int M, int N, int K, int relu){
    dim3 g(N/128, (M+63)/64);
    wmma_gemm_bf16<<<g, 256, 0, stream>>>(A, Bt, C0, bias, outF, outB, M, N, K, relu);
  };
  auto gb = [](int n){ return (n + 255) / 256; };

  // --- zero accumulators, edge counts ---
  zero_f32<<<gb(NR*NN), 256, 0, stream>>>(cnt, NR*NN);
  zero_f32<<<1, 4, 0, stream>>>(scal, 4);
  count_edges<<<gb(NE), 256, 0, stream>>>(edge_index, edge_type, cnt);

  // --- weight conversions (bf16, transposed to [N,K]) ---
  conv_f2b<<<gb(NN*DIN), 256, 0, stream>>>(x, xb, NN*DIN);
  convT_f2b<<<gb(DIN*DD), 256, 0, stream>>>(node_W, wtnode, DIN, DD);
  for (int l=0; l<NL; ++l){
    for (int r=0; r<NR; ++r)
      convT_f2b<<<gb(DD*DD), 256, 0, stream>>>(gcn_Wr + (size_t)(l*NR+r)*DD*DD,
                                               wtWr + (size_t)(l*NR+r)*DD*DD, DD, DD);
    convT_f2b<<<gb(DD*DD), 256, 0, stream>>>(gcn_Wroot + (size_t)l*DD*DD,
                                             wtWroot + (size_t)l*DD*DD, DD, DD);
  }
  convT_f2b<<<gb(DD*512), 256, 0, stream>>>(pu_W, wtpu, DD, 512);
  convT_f2b<<<gb(DD*512), 256, 0, stream>>>(qz_W, wtqz, DD, 512);
  convT_f2b<<<gb(512*DD), 256, 0, stream>>>(out_W1, wto1, 512, DD);

  // --- node transform: feats = x @ node_W + b ---
  gemm(xb, wtnode, nullptr, node_b, feats, featsb, NN, DD, DIN, 0);

  // --- RGCN layers ---
  for (int l=0; l<NL; ++l){
    zero_f32<<<gb(NN*DD), 256, 0, stream>>>(agg, NN*DD);
    for (int r=0; r<NR; ++r){
      gemm(featsb, wtWr + (size_t)(l*NR+r)*DD*DD, nullptr, nullptr, hrel, nullptr,
           NN, DD, DD, 0);
      scatter_rel<<<NE, DD, 0, stream>>>(edge_index, edge_type, cnt, hrel, agg, r);
    }
    // feats = agg + feats @ Wroot + b  (relu except last layer)
    gemm(featsb, wtWroot + (size_t)l*DD*DD, agg, gcn_b + (size_t)l*DD,
         feats, nullptr, NN, DD, DD, (l < NL-1) ? 1 : 0);
    conv_f2b<<<gb(NN*DD), 256, 0, stream>>>(feats, featsb, NN*DD);
  }

  // --- graph pooling ---
  zero_f32<<<gb(NM*DD), 256, 0, stream>>>(hm, NM*DD);
  zero_f32<<<gb(NM), 256, 0, stream>>>(gcnt, NM);
  pool_sum<<<NN, DD, 0, stream>>>(feats, batch, hm, gcnt);
  pool_div<<<gb(NM*DD), 256, 0, stream>>>(hm, gcnt);

  // --- H_all = [H_R; H_M], relu'd bf16 ---
  hall_relu<<<gb((NRAT+NM)*DD), 256, 0, stream>>>(H_R, hm, hallb);

  // --- pu / qz heads ---
  gemm(hallb, wtpu, nullptr, pu_b, pu_out, nullptr, NRAT+NM, 512, DD, 0);
  gemm(hallb, wtqz, nullptr, qz_b, qz_out, nullptr, NRAT+NM, 512, DD, 0);

  // --- samples ---
  sample_u<<<gb((NRAT+NM)*DD), 256, 0, stream>>>(pu_out, u, ku0, ku1);
  compute_An<<<NM, NRAT, 0, stream>>>(u, pg_ls, ka0, ka1, An);
  sample_z<<<gb(NM*DD), 256, 0, stream>>>(qz_out, zM, kzm0, kzm1, NM, NRAT);
  sample_z<<<gb(NRAT*DD), 256, 0, stream>>>(qz_out, zR, kzr0, kzr1, NRAT, 0);

  // --- VI term ---
  pz_logpqz<<<NM, DD, 0, stream>>>(An, qz_out, zM, &scal[3]);

  // --- prediction heads ---
  build_rep<<<gb(NM*512), 256, 0, stream>>>(zM, u, NRAT, repMb, NM);
  build_rep<<<gb(NRAT*512), 256, 0, stream>>>(zR, u, 0, repRb, NRAT);
  gemm(repMb, wto1, nullptr, out_b1, hidM, nullptr, NM, DD, 512, 1);
  gemm(repRb, wto1, nullptr, out_b1, hidR, nullptr, NRAT, DD, 512, 1);
  head_loss<<<NM, DD, 0, stream>>>(hidM, out_W2, out_b2, yM, &scal[0]);
  head_loss<<<NRAT, DD, 0, stream>>>(hidR, out_W2, out_b2, yR, &scal[1]);

  // --- distance regularizer ---
  reg_kernel<<<NM, 32, 0, stream>>>(u, yM, &scal[2]);

  // --- final scalar ---
  finalize<<<1, 1, 0, stream>>>(scal, out);
}
